// AtlasNetSphereEdgeConvGenerator_52527450030133
// MI455X (gfx1250) — matrix operand order repaired and checked by reading, not compiled, not run on hardware
//
#include <hip/hip_runtime.h>

typedef __attribute__((ext_vector_type(16))) __bf16 v16bf;
typedef __attribute__((ext_vector_type(8)))  float  v8f;
typedef unsigned short u16;
typedef unsigned int   u32;

#define NCOL 16384   // B*N = 8*2048
#define CDIM 512
#define NPTS 2048
#define KNN  8

union FragBF { v16bf v; u32 u[8]; };

__device__ __forceinline__ float lrelu(float x) { return x > 0.f ? x : 0.2f * x; }

__device__ __forceinline__ u16 f32_to_bf16(float f) {
  u32 u = __float_as_uint(f);
  u32 r = (u + 0x7FFFu + ((u >> 16) & 1u)) >> 16;  // round-to-nearest-even
  return (u16)r;
}

// ---------------- tiny prep kernels ----------------

__global__ void cvt_bf16_kernel(const float* __restrict__ s, u16* __restrict__ d, int n) {
  int i = blockIdx.x * blockDim.x + threadIdx.x;
  if (i < n) d[i] = f32_to_bf16(s[i]);
}

// Wz[b][c] = sum_j z[b][j] * W1[c][6+j]
__global__ void wz_kernel(const float* __restrict__ z, const float* __restrict__ W1,
                          float* __restrict__ Wz) {
  int i = blockIdx.x * blockDim.x + threadIdx.x;  // 4096 = 8*512
  int b = i >> 9, c = i & 511;
  const float* zr = z + b * 512;
  const float* wr = W1 + c * 518 + 6;
  float s = 0.f;
  for (int j = 0; j < 512; ++j) s += zr[j] * wr[j];
  Wz[i] = s;
}

// Fold BN into per-channel scale/shift applied to (Y + b4):
// lrelu( A4[c]*Y + B4[c] ) where A4 = g*rsqrt(v+eps), B4 = A4*b4 + bn_b - bn_m*A4
__global__ void bnfold_kernel(const float* __restrict__ b4, const float* __restrict__ g,
                              const float* __restrict__ bb, const float* __restrict__ m,
                              const float* __restrict__ v, float* __restrict__ A4,
                              float* __restrict__ B4) {
  int c = blockIdx.x * blockDim.x + threadIdx.x;
  if (c < 512) {
    float s = g[c] * rsqrtf(v[c] + 1e-5f);
    A4[c] = s;
    B4[c] = s * b4[c] + bb[c] - m[c] * s;
  }
}

// ---------------- layer 1 (geometry + z broadcast) ----------------
// X[col*512+c] = bf16( mean_k lrelu( Wz[b][c] + b1[c] + W1[c][0:3].cen + W1[c][3:6].rel_k ) )
__global__ __launch_bounds__(256) void layer1_kernel(
    const float* __restrict__ sphere, const int* __restrict__ knn,
    const float* __restrict__ W1, const float* __restrict__ b1,
    const float* __restrict__ Wz, u16* __restrict__ X) {
  int col = blockIdx.x;
  int b = col >> 11, n = col & 2047;
  __shared__ float cen[3];
  __shared__ float rel[KNN][3];
  int tid = threadIdx.x;
  if (tid < 3) cen[tid] = sphere[n * 3 + tid];
  if (tid < KNN) {
    int id = knn[n * KNN + tid];
    for (int j = 0; j < 3; ++j) rel[tid][j] = sphere[id * 3 + j] - sphere[n * 3 + j];
  }
  __syncthreads();
  for (int c = tid; c < CDIM; c += 256) {
    const float* w = W1 + c * 518;
    float base = Wz[b * 512 + c] + b1[c] + w[0] * cen[0] + w[1] * cen[1] + w[2] * cen[2];
    float acc = 0.f;
#pragma unroll
    for (int k = 0; k < KNN; ++k) {
      float vv = base + w[3] * rel[k][0] + w[4] * rel[k][1] + w[5] * rel[k][2];
      acc += lrelu(vv);
    }
    X[(size_t)col * CDIM + c] = f32_to_bf16(acc * 0.125f);
  }
}

// ---------------- WMMA GEMM:  Y[col*M+m] = sum_k A[m][k] * B[col][k] ----------------
// A: M x 512 row-major bf16 ; B: 16384 x 512 bf16 (col-major per column) ; K = 512
// Wave computes a 16(M) x 64(col) strip: 1 A-frag + 4 B-frags + 4 WMMAs per K-step.
__global__ __launch_bounds__(256) void gemm_bf16_kernel(
    const u16* __restrict__ A, const u16* __restrict__ B, float* __restrict__ Y, int M) {
  const int lane = threadIdx.x & 31;
  const int wave = threadIdx.x >> 5;
  const int mtiles = M >> 4;
  const int tile = blockIdx.x * 8 + wave;
  const int mt = tile % mtiles;
  const int ns = tile / mtiles;
  const int li = lane & 15;   // row within M-tile (A), col within N-tile (B/D)
  const int lh = lane >> 4;   // lane half
  const int m0 = mt << 4;
  const int c0 = ns << 6;     // 64 columns per wave

  v8f acc[4];
#pragma unroll
  for (int t = 0; t < 4; ++t) acc[t] = (v8f){0.f, 0.f, 0.f, 0.f, 0.f, 0.f, 0.f, 0.f};

  const u16* arow = A + (size_t)(m0 + li) * 512;
  for (int kb = 0; kb < 512; kb += 32) {
    FragBF a;
#pragma unroll
    for (int vv = 0; vv < 8; ++vv) {
      int k0 = ((vv >> 2) << 4) + (lh << 3) + ((vv & 3) << 1);  // A layout (ISA 7.12.2)
      a.u[vv] = *(const u32*)(arow + kb + k0);
    }
#pragma unroll
    for (int t = 0; t < 4; ++t) {
      const u16* bcol = B + (size_t)(c0 + t * 16 + li) * 512 + kb;
      FragBF bf;
#pragma unroll
      for (int vv = 0; vv < 8; ++vv) {
        int k0 = ((vv >> 2) << 4) + (lh << 3) + ((vv & 3) << 1);  // B layout (same k order as A)
        bf.u[vv] = *(const u32*)(bcol + k0);
      }
      acc[t] = __builtin_amdgcn_wmma_f32_16x16x32_bf16(
          false, a.v, false, bf.v, (short)0, acc[t], false, false);
    }
  }
  // D layout: elem r -> m = m0 + 8*lh + r, n = li  -> 8 contiguous floats per lane
#pragma unroll
  for (int t = 0; t < 4; ++t) {
    float* yc = Y + (size_t)(c0 + t * 16 + li) * M + m0 + (lh << 3);
#pragma unroll
    for (int r = 0; r < 8; ++r) yc[r] = acc[t][r];
  }
}

// ---------------- gather + activation + mean over K neighbors ----------------
// out(b,c,n) = mean_k lrelu( Ac[c]*Y[(b*2048+idx[n][k])*512+c] + Bc[c] )
__global__ __launch_bounds__(256) void gather_act_kernel(
    const float* __restrict__ Y, const int* __restrict__ knn,
    const float* __restrict__ Ac, const float* __restrict__ Bc,
    u16* __restrict__ Xout, float* __restrict__ Fout) {
  int col = blockIdx.x;
  int b = col >> 11, n = col & 2047;
  __shared__ int idx[KNN];
  if (threadIdx.x < KNN) idx[threadIdx.x] = knn[n * KNN + threadIdx.x];
  __syncthreads();
  size_t base_b = (size_t)b * NPTS;
  for (int c = threadIdx.x; c < CDIM; c += 256) {
    float a = Ac ? Ac[c] : 1.0f;
    float bb = Bc[c];
    float acc = 0.f;
#pragma unroll
    for (int k = 0; k < KNN; ++k) {
      float vv = a * Y[(base_b + (size_t)idx[k]) * CDIM + c] + bb;
      acc += lrelu(vv);
    }
    acc *= 0.125f;
    if (Xout) Xout[(size_t)col * CDIM + c] = f32_to_bf16(acc);
    if (Fout) Fout[(size_t)b * (CDIM * NPTS) + (size_t)c * NPTS + n] = acc;  // (B,512,N)
  }
}

// ---------------- final head: lrelu(t)=lrelu(Yt+bx1); pcs = sigmoid(Wx2 t + bx2)-0.5 ----
__global__ __launch_bounds__(256) void head_kernel(
    const float* __restrict__ Yt, const float* __restrict__ bx1,
    const float* __restrict__ Wx2, const float* __restrict__ bx2,
    float* __restrict__ pcs) {
  int col = blockIdx.x * blockDim.x + threadIdx.x;  // 16384
  int b = col >> 11, n = col & 2047;
  const float* y = Yt + (size_t)col * 64;
  float t[64];
#pragma unroll
  for (int j = 0; j < 64; ++j) t[j] = lrelu(y[j] + bx1[j]);
#pragma unroll
  for (int o = 0; o < 3; ++o) {
    float s = bx2[o];
#pragma unroll
    for (int j = 0; j < 64; ++j) s += Wx2[o * 64 + j] * t[j];
    float p = 1.f / (1.f + __expf(-s)) - 0.5f;
    pcs[(size_t)b * (NPTS * 3) + (size_t)n * 3 + o] = p;  // (B,N,3)
  }
}

// ---------------- driver ----------------
extern "C" void kernel_launch(void* const* d_in, const int* in_sizes, int n_in,
                              void* d_out, int out_size, void* d_ws, size_t ws_size,
                              hipStream_t stream) {
  (void)in_sizes; (void)n_in; (void)out_size; (void)ws_size;
  const float* z      = (const float*)d_in[0];
  const float* sphere = (const float*)d_in[1];
  const int*   knn    = (const int*)d_in[2];
  const float* W1     = (const float*)d_in[3];
  const float* b1     = (const float*)d_in[4];
  const float* W2     = (const float*)d_in[5];
  const float* b2     = (const float*)d_in[6];
  const float* W3     = (const float*)d_in[7];
  const float* b3     = (const float*)d_in[8];
  const float* W4     = (const float*)d_in[9];
  const float* b4     = (const float*)d_in[10];
  const float* bn_g   = (const float*)d_in[11];
  const float* bn_b   = (const float*)d_in[12];
  const float* bn_m   = (const float*)d_in[13];
  const float* bn_v   = (const float*)d_in[14];
  const float* Wx1    = (const float*)d_in[15];
  const float* bx1    = (const float*)d_in[16];
  const float* Wx2    = (const float*)d_in[17];
  const float* bx2    = (const float*)d_in[18];
  float* out = (float*)d_out;  // [pcs: 8*2048*3][feature: 8*512*2048]

  char* ws = (char*)d_ws;
  // workspace layout (bytes)
  u16*  Xbuf0 = (u16*)(ws + 0);                    // 16384*512*2 = 16 MB
  u16*  Xbuf1 = (u16*)(ws + 16777216);             // 16 MB
  float* Ybuf = (float*)(ws + 33554432);           // 16384*512*4 = 32 MB
  u16*  Wbf2  = (u16*)(ws + 67108864);             // 512 KB
  u16*  Wbf3  = (u16*)(ws + 67633152);
  u16*  Wbf4  = (u16*)(ws + 68157440);
  u16*  Wx1bf = (u16*)(ws + 68681728);             // 64 KB
  float* Wz   = (float*)(ws + 68747264);           // 16 KB
  float* A4   = (float*)(ws + 68763648);           // 2 KB
  float* B4   = (float*)(ws + 68765696);           // 2 KB
  float* Yt   = Ybuf;                              // head GEMM output reuses Ybuf (4 MB)

  // prep: bf16 weight copies + z-projection + BN fold
  cvt_bf16_kernel<<<1024, 256, 0, stream>>>(W2, Wbf2, 512 * 512);
  cvt_bf16_kernel<<<1024, 256, 0, stream>>>(W3, Wbf3, 512 * 512);
  cvt_bf16_kernel<<<1024, 256, 0, stream>>>(W4, Wbf4, 512 * 512);
  cvt_bf16_kernel<<<128, 256, 0, stream>>>(Wx1, Wx1bf, 64 * 512);
  wz_kernel<<<16, 256, 0, stream>>>(z, W1, Wz);
  bnfold_kernel<<<2, 256, 0, stream>>>(b4, bn_g, bn_b, bn_m, bn_v, A4, B4);

  // layer 1 (geometry + z broadcast, gather-free form)
  layer1_kernel<<<NCOL, 256, 0, stream>>>(sphere, knn, W1, b1, Wz, Xbuf0);

  // layers 2..4: GEMM then gathered activation-mean (conv commutes with gather)
  gemm_bf16_kernel<<<1024, 256, 0, stream>>>(Wbf2, Xbuf0, Ybuf, 512);
  gather_act_kernel<<<NCOL, 256, 0, stream>>>(Ybuf, knn, nullptr, b2, Xbuf1, nullptr);

  gemm_bf16_kernel<<<1024, 256, 0, stream>>>(Wbf3, Xbuf1, Ybuf, 512);
  gather_act_kernel<<<NCOL, 256, 0, stream>>>(Ybuf, knn, nullptr, b3, Xbuf0, nullptr);

  gemm_bf16_kernel<<<1024, 256, 0, stream>>>(Wbf4, Xbuf0, Ybuf, 512);
  // layer 4: BN folded in; writes feature (f32, second output) + bf16 copy for head
  gather_act_kernel<<<NCOL, 256, 0, stream>>>(Ybuf, knn, A4, B4, Xbuf1,
                                              out + (size_t)8 * 2048 * 3);

  // head: t = Wx1 @ feature  (M=64 GEMM, same WMMA kernel), then sigmoid head
  gemm_bf16_kernel<<<128, 256, 0, stream>>>(Wx1bf, Xbuf1, Yt, 64);
  head_kernel<<<64, 256, 0, stream>>>(Yt, bx1, Wx2, bx2, out);
}
